// RefMLA_2911987827146
// MI455X (gfx1250) — matrix operand order repaired and checked by reading, not compiled, not run on hardware
//
#include <hip/hip_runtime.h>
#include <math.h>

// ---------------------------------------------------------------------------
// DeepSeek-style sparse MLA block for MI455X (gfx1250, wave32, WMMA bf16,
// TDM async tile staging into LDS for the large GEMMs).
// ---------------------------------------------------------------------------

constexpr int S  = 2048, DM = 2048, QR = 1536;
constexpr int H  = 16,  DN = 128, DR = 64, DV = 128, KR = 512;
constexpr int HI = 32,  DI = 128, TOPK = 512;
constexpr int DQK = DN + DR;                       // 192
constexpr float SC_ATT = 0.07216878364870323f;     // 1/sqrt(192)
constexpr float SC_IDX = 0.08838834764831845f;     // 1/sqrt(128)
constexpr float SC_WTS = 0.17677669529663687f;     // 1/sqrt(32)

typedef __attribute__((ext_vector_type(16))) __bf16 v16bf;
typedef __attribute__((ext_vector_type(8)))  float  v8f;

#define DEVFN __device__ __forceinline__

#if defined(__has_builtin)
#if __has_builtin(__builtin_amdgcn_tensor_load_to_lds) && \
    __has_builtin(__builtin_amdgcn_s_wait_tensorcnt)
#define HAVE_TDM 1
#endif
#endif

DEVFN v8f wmma_bf16(v16bf a, v16bf b, v8f c) {
  // D = A(16x32 bf16) * B(32x16 bf16) + C(16x16 f32)
  return __builtin_amdgcn_wmma_f32_16x16x32_bf16(false, a, false, b,
                                                 (short)0, c, false, false);
}

// Load 16 bf16 halves: [0..7] from p, [8..15] from p+step (both 16B aligned).
DEVFN v16bf load_frag16(const __bf16* p, int step) {
  v16bf r;
  ((uint4*)&r)[0] = *(const uint4*)p;
  ((uint4*)&r)[1] = *(const uint4*)(p + step);
  return r;
}

#if HAVE_TDM
typedef __attribute__((ext_vector_type(4))) unsigned tdm_v4u;
typedef __attribute__((ext_vector_type(8))) int      tdm_v8i;
typedef __attribute__((ext_vector_type(4))) int      tdm_v4i;

// Issue one TDM 2D tile load: 128 rows x 32 bf16 (64B) from a row-major
// [rows x ldk] bf16 tensor into LDS (rows contiguous, 64B per row).
// D# packing per CDNA5 ISA §8.3/8.4 (group0: count/lds/global/type,
// group1: data_size=2B, tensor dims, tile 128x32, dim0 stride).
DEVFN void tdm_issue_tile(const __bf16* g, const __bf16* lds_ptr,
                          unsigned tensor_rows, unsigned ldk) {
  typedef __attribute__((address_space(3))) const void* lds_cptr;
  unsigned ldsoff = (unsigned)(unsigned long long)(lds_cptr)lds_ptr;
  unsigned long long ga = (unsigned long long)g;
  tdm_v4u g0;
  g0[0] = 1u;                                          // count=1 (valid user D#)
  g0[1] = ldsoff;                                      // lds_addr (bytes)
  g0[2] = (unsigned)(ga & 0xFFFFFFFFu);                // global_addr[31:0]
  g0[3] = (unsigned)((ga >> 32) & 0x01FFFFFFu) | (2u << 30);  // [56:32] | type=2
  tdm_v8i g1;
  g1[0] = (int)(1u << 16);                             // data_size = 2 bytes
  g1[1] = (int)((ldk & 0xFFFFu) << 16);                // tensor_dim0[15:0]
  g1[2] = (int)(((ldk >> 16) & 0xFFFFu) |
                ((tensor_rows & 0xFFFFu) << 16));      // dim0 hi | tensor_dim1 lo
  g1[3] = (int)(((tensor_rows >> 16) & 0xFFFFu) |
                (32u << 16));                          // dim1 hi | tile_dim0=32
  g1[4] = 128;                                         // tile_dim1=128 (dim2=0)
  g1[5] = (int)ldk;                                    // tensor_dim0_stride lo32
  g1[6] = 0;                                           // stride hi | dim1_stride lo
  g1[7] = 0;
  tdm_v4i z4 = (tdm_v4i)0;
#if __clang_major__ >= 23
  tdm_v8i z8 = (tdm_v8i)0;
  __builtin_amdgcn_tensor_load_to_lds(g0, g1, z4, z4, z8, 0);
#else
  __builtin_amdgcn_tensor_load_to_lds(g0, g1, z4, z4, 0);
#endif
}
#endif  // HAVE_TDM

// ---------------------------------------------------------------------------
// f32 -> bf16 conversion
// ---------------------------------------------------------------------------
__global__ void cvt_bf16_kernel(const float* __restrict__ in,
                                __bf16* __restrict__ out, size_t n) {
  size_t i = (size_t)blockIdx.x * blockDim.x + threadIdx.x;
  size_t st = (size_t)gridDim.x * blockDim.x;
  for (; i < n; i += st) out[i] = (__bf16)in[i];
}

// ---------------------------------------------------------------------------
// LDS-tiled WMMA GEMM with TDM double buffering:
//   C[M,N] (f32 * scale) = A[M,K] bf16  x  B[N,K] bf16 ^T
// Block = 256 threads (8 waves) -> 128x128 C tile; wave w owns rows w*16..+15.
// K-step 32: 8KB A tile + 8KB B tile per buffer, two buffers.
// Requires M%128==0, N%128==0, K%32==0.
// ---------------------------------------------------------------------------
__global__ __launch_bounds__(256) void gemm_lds_kernel(
    const __bf16* __restrict__ A, const __bf16* __restrict__ B,
    float* __restrict__ C, int M, int N, int K, float scale) {
  __shared__ __align__(16) __bf16 sA[2][128 * 32];
  __shared__ __align__(16) __bf16 sB[2][128 * 32];
  const int lane = threadIdx.x & 31;
  const int wave = threadIdx.x >> 5;
  const int bn = N / 128;
  const int m0 = (blockIdx.x / bn) * 128;
  const int n0 = (blockIdx.x % bn) * 128;
  const int m = lane & 15, hl = lane >> 4;

  v8f acc[8];
#pragma unroll
  for (int j = 0; j < 8; ++j) acc[j] = (v8f)0.0f;

  const int KT = K / 32;

#if HAVE_TDM
  if (wave == 0) {  // TDM ignores EXEC; wave-level issue of the first tile pair
    tdm_issue_tile(A + (size_t)m0 * K, &sA[0][0], (unsigned)M, (unsigned)K);
    tdm_issue_tile(B + (size_t)n0 * K, &sB[0][0], (unsigned)N, (unsigned)K);
  }
#endif

  for (int kt = 0; kt < KT; ++kt) {
    const int cur = kt & 1;
    const int k0 = kt * 32;
    __syncthreads();  // all waves done reading the buffer about to be refilled
#if HAVE_TDM
    if (wave == 0) {
      if (kt + 1 < KT) {
        const int nxt = (kt + 1) & 1, kn = (kt + 1) * 32;
        tdm_issue_tile(A + (size_t)m0 * K + kn, &sA[nxt][0], (unsigned)M, (unsigned)K);
        tdm_issue_tile(B + (size_t)n0 * K + kn, &sB[nxt][0], (unsigned)N, (unsigned)K);
        __builtin_amdgcn_s_wait_tensorcnt(2);  // pair(kt) complete, pair(kt+1) inflight
      } else {
        __builtin_amdgcn_s_wait_tensorcnt(0);
      }
    }
#else
    // cooperative fallback: 512 x 16B chunks per tile
    for (int c = threadIdx.x; c < 512; c += 256) {
      int row = c >> 2, k8 = (c & 3) * 8;
      *(uint4*)&sA[cur][row * 32 + k8] =
          *(const uint4*)(A + (size_t)(m0 + row) * K + k0 + k8);
      *(uint4*)&sB[cur][row * 32 + k8] =
          *(const uint4*)(B + (size_t)(n0 + row) * K + k0 + k8);
    }
#endif
    __syncthreads();  // tile(kt) visible to all waves

    v16bf a = load_frag16(&sA[cur][(wave * 16 + m) * 32 + hl * 16], 8);
#pragma unroll
    for (int j = 0; j < 8; ++j) {
      v16bf b = load_frag16(&sB[cur][(j * 16 + m) * 32 + hl * 16], 8);
      acc[j] = wmma_bf16(a, b, acc[j]);
    }
  }

#pragma unroll
  for (int j = 0; j < 8; ++j)
#pragma unroll
    for (int r = 0; r < 8; ++r)
      C[(size_t)(m0 + wave * 16 + r + hl * 8) * N + n0 + j * 16 + m] =
          acc[j][r] * scale;
}

// ---------------------------------------------------------------------------
// Direct (register-fragment) WMMA GEMM for shapes not divisible by 128.
// One wave computes a 16 x (16*NT) tile. 8 waves / block.
// ---------------------------------------------------------------------------
template <int NT>
__global__ __launch_bounds__(256) void gemm_bt_kernel(
    const __bf16* __restrict__ A, const __bf16* __restrict__ B,
    float* __restrict__ C, int M, int N, int K, float scale) {
  const int lane = threadIdx.x & 31;
  const int wave = threadIdx.x >> 5;
  const int gwave = blockIdx.x * 8 + wave;
  const int tilesN = N / (16 * NT);
  const int total = (M / 16) * tilesN;
  if (gwave >= total) return;
  const int m0 = (gwave / tilesN) * 16;
  const int n0 = (gwave % tilesN) * (16 * NT);

  const int m = lane & 15, hl = lane >> 4;
  v8f acc[NT];
#pragma unroll
  for (int j = 0; j < NT; ++j) acc[j] = (v8f)0.0f;

  const __bf16* arow = A + (size_t)(m0 + m) * K + hl * 16;
  for (int k0 = 0; k0 < K; k0 += 32) {
    __builtin_prefetch(arow + k0 + 128, 0, 1);          // global_prefetch_b8
    v16bf a = load_frag16(arow + k0, 8);
#pragma unroll
    for (int j = 0; j < NT; ++j) {
      const __bf16* brow = B + (size_t)(n0 + j * 16 + m) * K + k0 + hl * 16;
      v16bf b = load_frag16(brow, 8);
      acc[j] = wmma_bf16(a, b, acc[j]);
    }
  }
#pragma unroll
  for (int j = 0; j < NT; ++j)
#pragma unroll
    for (int r = 0; r < 8; ++r)
      C[(size_t)(m0 + r + hl * 8) * N + n0 + j * 16 + m] = acc[j][r] * scale;
}

// ---------------------------------------------------------------------------
// RMSNorm over a row slice: y_bf16[row, 0:cols] from x_f32[row*ldx + 0:cols]
// ---------------------------------------------------------------------------
__global__ __launch_bounds__(256) void rmsnorm_kernel(
    const float* __restrict__ x, const float* __restrict__ w,
    __bf16* __restrict__ y, int cols, int ldx, int ldy) {
  __shared__ float red[8];
  __shared__ float rn_s;
  const int row = blockIdx.x;
  const float* xr = x + (size_t)row * ldx;
  float ss = 0.f;
  for (int c = threadIdx.x; c < cols; c += blockDim.x) { float v = xr[c]; ss += v * v; }
  for (int msk = 16; msk >= 1; msk >>= 1) ss += __shfl_xor(ss, msk, 32);
  if ((threadIdx.x & 31) == 0) red[threadIdx.x >> 5] = ss;
  __syncthreads();
  if (threadIdx.x == 0) {
    float t = 0.f;
    for (int i = 0; i < 8; ++i) t += red[i];
    rn_s = rsqrtf(t / (float)cols + 1e-6f);
  }
  __syncthreads();
  const float rn = rn_s;
  __bf16* yr = y + (size_t)row * ldy;
  for (int c = threadIdx.x; c < cols; c += blockDim.x)
    yr[c] = (__bf16)(xr[c] * rn * w[c]);
}

// ---------------------------------------------------------------------------
// q rope (interleaved on dims 128..191 per head) + cast to bf16
// ---------------------------------------------------------------------------
__global__ void rope_q_kernel(const float* __restrict__ q,
                              const float* __restrict__ cf,
                              const float* __restrict__ sf,
                              __bf16* __restrict__ out) {
  const size_t n = (size_t)S * H * DQK;
  size_t i = (size_t)blockIdx.x * blockDim.x + threadIdx.x;
  size_t st = (size_t)gridDim.x * blockDim.x;
  for (; i < n; i += st) {
    int d = (int)(i % DQK);
    size_t sh = i / DQK;
    int s = (int)(sh / H);
    float val;
    if (d < DN) {
      val = q[i];
    } else {
      int j = d - DN, p = j >> 1;
      float c = cf[(size_t)s * 32 + p], sn = sf[(size_t)s * 32 + p];
      size_t base = sh * DQK + DN + 2 * p;
      float x0 = q[base], x1 = q[base + 1];
      val = (j & 1) ? (x0 * sn + x1 * c) : (x0 * c - x1 * sn);
    }
    out[i] = (__bf16)val;
  }
}

// ---------------------------------------------------------------------------
// k_pe: interleaved rope of kv_all[:, 512:576] -> kpe f32 [S,64]
// ---------------------------------------------------------------------------
__global__ void kpe_kernel(const float* __restrict__ kvall,
                           const float* __restrict__ cf,
                           const float* __restrict__ sf,
                           float* __restrict__ kpe) {
  const size_t n = (size_t)S * 64;
  size_t i = (size_t)blockIdx.x * blockDim.x + threadIdx.x;
  size_t st = (size_t)gridDim.x * blockDim.x;
  for (; i < n; i += st) {
    int j = (int)(i & 63);
    int s = (int)(i >> 6);
    int p = j >> 1;
    float c = cf[(size_t)s * 32 + p], sn = sf[(size_t)s * 32 + p];
    const float* base = kvall + (size_t)s * (KR + DR) + KR + 2 * p;
    float x0 = base[0], x1 = base[1];
    kpe[i] = (j & 1) ? (x0 * sn + x1 * c) : (x0 * c - x1 * sn);
  }
}

// ---------------------------------------------------------------------------
// Assemble k_bf [S,H,192] (nope||pe) and v transposed vt_bf [H*128, S]
// ---------------------------------------------------------------------------
__global__ void assemble_kv_kernel(const float* __restrict__ kvp,
                                   const float* __restrict__ kpe,
                                   __bf16* __restrict__ kb,
                                   __bf16* __restrict__ vtb) {
  const size_t n = (size_t)S * H * 320;
  size_t i = (size_t)blockIdx.x * blockDim.x + threadIdx.x;
  size_t st = (size_t)gridDim.x * blockDim.x;
  for (; i < n; i += st) {
    int e = (int)(i % 320);
    size_t sh = i / 320;
    int hh = (int)(sh % H);
    int s = (int)(sh / H);
    if (e < 128) {
      kb[(size_t)s * (H * DQK) + hh * DQK + e] =
          (__bf16)kvp[(size_t)s * (H * 256) + hh * 256 + e];
    } else if (e < 256) {
      int d = e - 128;
      vtb[((size_t)(hh * DV + d)) * S + s] =
          (__bf16)kvp[(size_t)s * (H * 256) + hh * 256 + e];
    } else {
      int j = e - 256;
      kb[(size_t)s * (H * DQK) + hh * DQK + DN + j] =
          (__bf16)kpe[(size_t)s * 64 + j];
    }
  }
}

// ---------------------------------------------------------------------------
// qi rope_half (on dims 0..63 of each 128-wide indexer head) + bf16 cast
// ---------------------------------------------------------------------------
__global__ void rope_qi_kernel(const float* __restrict__ qi,
                               const float* __restrict__ cf,
                               const float* __restrict__ sf,
                               __bf16* __restrict__ out) {
  const size_t n = (size_t)S * HI * DI;
  size_t i = (size_t)blockIdx.x * blockDim.x + threadIdx.x;
  size_t st = (size_t)gridDim.x * blockDim.x;
  for (; i < n; i += st) {
    int d = (int)(i % DI);
    size_t sh = i / DI;
    int s = (int)(sh / HI);
    float val;
    if (d < 32) {
      float c = cf[(size_t)s * 32 + d], sn = sf[(size_t)s * 32 + d];
      val = qi[sh * DI + d] * c - qi[sh * DI + d + 32] * sn;
    } else if (d < 64) {
      int p = d - 32;
      float c = cf[(size_t)s * 32 + p], sn = sf[(size_t)s * 32 + p];
      val = qi[sh * DI + p] * sn + qi[sh * DI + d] * c;
    } else {
      val = qi[i];
    }
    out[i] = (__bf16)val;
  }
}

// ---------------------------------------------------------------------------
// ki: layernorm(128) then rope_half on first 64, cast to bf16. One row/block.
// ---------------------------------------------------------------------------
__global__ __launch_bounds__(128) void ki_ln_rope_kernel(
    const float* __restrict__ x, const float* __restrict__ w,
    const float* __restrict__ b, const float* __restrict__ cf,
    const float* __restrict__ sf, __bf16* __restrict__ y) {
  __shared__ float red[4];
  __shared__ float m_s, r_s;
  __shared__ float v[128];
  const int row = blockIdx.x, c = threadIdx.x;
  float xv = x[(size_t)row * DI + c];
  float t = xv;
  for (int msk = 16; msk >= 1; msk >>= 1) t += __shfl_xor(t, msk, 32);
  if ((c & 31) == 0) red[c >> 5] = t;
  __syncthreads();
  if (c == 0) m_s = (red[0] + red[1] + red[2] + red[3]) * (1.0f / 128.0f);
  __syncthreads();
  float d = xv - m_s;
  t = d * d;
  for (int msk = 16; msk >= 1; msk >>= 1) t += __shfl_xor(t, msk, 32);
  if ((c & 31) == 0) red[c >> 5] = t;
  __syncthreads();
  if (c == 0) r_s = rsqrtf((red[0] + red[1] + red[2] + red[3]) * (1.0f / 128.0f) + 1e-5f);
  __syncthreads();
  float nv = d * r_s * w[c] + b[c];
  v[c] = nv;
  __syncthreads();
  float outv;
  if (c < 32) {
    float cc = cf[(size_t)row * 32 + c], ss = sf[(size_t)row * 32 + c];
    outv = v[c] * cc - v[c + 32] * ss;
  } else if (c < 64) {
    int p = c - 32;
    float cc = cf[(size_t)row * 32 + p], ss = sf[(size_t)row * 32 + p];
    outv = v[p] * ss + v[c] * cc;
  } else {
    outv = nv;
  }
  y[(size_t)row * DI + c] = (__bf16)outv;
}

// ---------------------------------------------------------------------------
// Indexer scores: per 16x16 (s,t) tile, sum over 32 heads of
// relu(qi.ki / sqrt(DI)) * wts[s,head]; causal -inf. 128 WMMAs / tile.
// ---------------------------------------------------------------------------
__global__ __launch_bounds__(256) void idx_scores_kernel(
    const __bf16* __restrict__ qi, const __bf16* __restrict__ ki,
    const float* __restrict__ wts, float* __restrict__ out) {
  const int lane = threadIdx.x & 31;
  const int wave = threadIdx.x >> 5;
  const int tile = blockIdx.x * 8 + wave;
  const int s0 = (tile / (S / 16)) * 16;
  const int t0 = (tile % (S / 16)) * 16;
  const int n = lane & 15, hl = lane >> 4;
  if (t0 > s0 + 15) {               // fully above the diagonal
#pragma unroll
    for (int r = 0; r < 8; ++r)
      out[(size_t)(s0 + r + hl * 8) * S + t0 + n] = -INFINITY;
    return;
  }
  float tot[8] = {0.f, 0.f, 0.f, 0.f, 0.f, 0.f, 0.f, 0.f};
  const __bf16* qrow = qi + (size_t)(s0 + n) * (HI * DI) + hl * 16;
  const __bf16* krow = ki + (size_t)(t0 + n) * DI + hl * 16;
  for (int hi = 0; hi < HI; ++hi) {
    v8f sc = (v8f)0.0f;
#pragma unroll
    for (int k0 = 0; k0 < DI; k0 += 32) {
      v16bf a = load_frag16(qrow + hi * DI + k0, 8);
      v16bf b = load_frag16(krow + k0, 8);
      sc = wmma_bf16(a, b, sc);
    }
#pragma unroll
    for (int r = 0; r < 8; ++r) {
      float wv = wts[(size_t)(s0 + r + hl * 8) * HI + hi];
      tot[r] += fmaxf(sc[r] * SC_IDX, 0.f) * wv;
    }
  }
#pragma unroll
  for (int r = 0; r < 8; ++r) {
    int srow = s0 + r + hl * 8, tcol = t0 + n;
    out[(size_t)srow * S + tcol] = (tcol <= srow) ? tot[r] : -INFINITY;
  }
}

// ---------------------------------------------------------------------------
// Per-row top-512 via 32-bit radix select -> selection bitmask (64 words/row)
// ---------------------------------------------------------------------------
__global__ __launch_bounds__(256) void topk_mask_kernel(
    const float* __restrict__ scores, unsigned* __restrict__ maskw) {
  __shared__ int redi[8];
  __shared__ int tot_s;
  __shared__ int tie_ctr;
  __shared__ unsigned words[64];
  const int row = blockIdx.x;
  const float* r = scores + (size_t)row * S;
  unsigned u[8];
#pragma unroll
  for (int i = 0; i < 8; ++i) {
    unsigned b = __float_as_uint(r[threadIdx.x + i * 256]);
    u[i] = (b & 0x80000000u) ? ~b : (b | 0x80000000u);  // order-preserving map
  }
  unsigned prefix = 0;
  for (int bit = 31; bit >= 0; --bit) {
    unsigned cand = prefix | (1u << bit);
    int cnt = 0;
#pragma unroll
    for (int i = 0; i < 8; ++i) cnt += (u[i] >= cand);
    for (int msk = 16; msk >= 1; msk >>= 1) cnt += __shfl_xor(cnt, msk, 32);
    if ((threadIdx.x & 31) == 0) redi[threadIdx.x >> 5] = cnt;
    __syncthreads();
    if (threadIdx.x == 0) {
      int t = 0;
      for (int j = 0; j < 8; ++j) t += redi[j];
      tot_s = t;
    }
    __syncthreads();
    if (tot_s >= TOPK) prefix = cand;
    __syncthreads();
  }
  // count strictly-greater, remaining slots go to ties
  int gt = 0;
#pragma unroll
  for (int i = 0; i < 8; ++i) gt += (u[i] > prefix);
  for (int msk = 16; msk >= 1; msk >>= 1) gt += __shfl_xor(gt, msk, 32);
  if ((threadIdx.x & 31) == 0) redi[threadIdx.x >> 5] = gt;
  __syncthreads();
  if (threadIdx.x == 0) {
    int t = 0;
    for (int j = 0; j < 8; ++j) t += redi[j];
    tie_ctr = TOPK - t;
  }
  if (threadIdx.x < 64) words[threadIdx.x] = 0u;
  __syncthreads();
#pragma unroll
  for (int i = 0; i < 8; ++i) {
    int t = threadIdx.x + i * 256;
    bool sel = (u[i] > prefix);
    if (!sel && u[i] == prefix) sel = (atomicSub(&tie_ctr, 1) > 0);
    if (sel) atomicOr(&words[t >> 5], 1u << (t & 31));
  }
  __syncthreads();
  if (threadIdx.x < 64) maskw[(size_t)row * 64 + threadIdx.x] = words[threadIdx.x];
}

// ---------------------------------------------------------------------------
// Sparse flash attention: one wave = (16 q rows, 1 head). Online softmax over
// selected+causal columns; probs go through LDS to become WMMA A-fragments;
// V is pre-transposed so PV B-fragments are contiguous loads.
// ---------------------------------------------------------------------------
__global__ __launch_bounds__(512) void attn_kernel(
    const __bf16* __restrict__ qb, const __bf16* __restrict__ kb,
    const __bf16* __restrict__ vt, const unsigned* __restrict__ maskw,
    __bf16* __restrict__ outb) {
  __shared__ __align__(16) __bf16 plds[16][512];   // per-wave 16x32 tile
  const int lane = threadIdx.x & 31;
  const int h = threadIdx.x >> 5;                  // head = wave id (16 waves)
  const int s0 = blockIdx.x * 16;
  const int n = lane & 15, hl = lane >> 4;

  v8f o[8];
#pragma unroll
  for (int j = 0; j < 8; ++j) o[j] = (v8f)0.0f;
  float mrow[8], lrow[8];
#pragma unroll
  for (int r = 0; r < 8; ++r) { mrow[r] = -INFINITY; lrow[r] = 0.f; }

  const __bf16* qrow = qb + (size_t)(s0 + n) * (H * DQK) + h * DQK + hl * 16;
  __bf16* pl = &plds[h][0];

  for (int t0 = 0; t0 <= s0 + 15; t0 += 32) {
    // skip tiles with no selected column for any of the 16 rows
    unsigned wv = (lane < 16) ? maskw[(size_t)(s0 + lane) * 64 + (t0 >> 5)] : 0u;
    if (!__any(wv != 0u)) continue;

    // ---- scores: q(16x192) . k^T -> two 16x16 tiles (t0..t0+15, +16..31)
    v8f sc0 = (v8f)0.0f, sc1 = (v8f)0.0f;
#pragma unroll
    for (int k0 = 0; k0 < DQK; k0 += 32) {
      v16bf a = load_frag16(qrow + k0, 8);
      const __bf16* kr0 = kb + (size_t)(t0 + n) * (H * DQK) + h * DQK + k0 + hl * 16;
      const __bf16* kr1 = kb + (size_t)(t0 + 16 + n) * (H * DQK) + h * DQK + k0 + hl * 16;
      v16bf b0 = load_frag16(kr0, 8);
      v16bf b1 = load_frag16(kr1, 8);
      sc0 = wmma_bf16(a, b0, sc0);
      sc1 = wmma_bf16(a, b1, sc1);
    }

    // ---- masking + online softmax (per row r of this lane's half)
#pragma unroll
    for (int r = 0; r < 8; ++r) {
      const int srow = s0 + r + hl * 8;
      const unsigned wr = maskw[(size_t)srow * 64 + (t0 >> 5)];
      const int tn0 = t0 + n, tn1 = t0 + 16 + n;
      float a0 = ((tn0 <= srow) && ((wr >> (tn0 & 31)) & 1u)) ? sc0[r] * SC_ATT : -INFINITY;
      float a1 = ((tn1 <= srow) && ((wr >> (tn1 & 31)) & 1u)) ? sc1[r] * SC_ATT : -INFINITY;
      float mx = fmaxf(a0, a1);
      for (int msk = 8; msk >= 1; msk >>= 1) mx = fmaxf(mx, __shfl_xor(mx, msk, 32));
      float mnew = fmaxf(mrow[r], mx);
      float alpha, e0, e1;
      if (mnew == -INFINITY) {
        alpha = 1.f; e0 = 0.f; e1 = 0.f;
      } else {
        alpha = (mrow[r] == -INFINITY) ? 0.f : __expf(mrow[r] - mnew);
        e0 = (a0 == -INFINITY) ? 0.f : __expf(a0 - mnew);
        e1 = (a1 == -INFINITY) ? 0.f : __expf(a1 - mnew);
      }
      float rs = e0 + e1;
      for (int msk = 8; msk >= 1; msk >>= 1) rs += __shfl_xor(rs, msk, 32);
      lrow[r] = lrow[r] * alpha + rs;
      mrow[r] = mnew;
#pragma unroll
      for (int j = 0; j < 8; ++j) o[j][r] *= alpha;
      // stash probs in LDS (16x32 row-major, this wave's private slab)
      pl[(r + hl * 8) * 32 + n] = (__bf16)e0;
      pl[(r + hl * 8) * 32 + 16 + n] = (__bf16)e1;
    }

    // ---- reload probs as an A-fragment (16x32)
    v16bf pa = load_frag16(pl + n * 32 + hl * 16, 8);

    // ---- out += probs(16x32) x V(32x128): 8 WMMAs against transposed V
#pragma unroll
    for (int j = 0; j < 8; ++j) {
      const __bf16* vr = vt + ((size_t)(h * DV + j * 16 + n)) * S + t0 + hl * 16;
      v16bf vb = load_frag16(vr, 8);
      o[j] = wmma_bf16(pa, vb, o[j]);
    }
  }

  // ---- finalize: divide by softmax denominator, store bf16
#pragma unroll
  for (int j = 0; j < 8; ++j)
#pragma unroll
    for (int r = 0; r < 8; ++r) {
      float val = (lrow[r] > 0.f) ? o[j][r] / lrow[r] : 0.f;
      outb[(size_t)(s0 + r + hl * 8) * (H * DV) + h * DV + j * 16 + n] = (__bf16)val;
    }
}

// ---------------------------------------------------------------------------
// Host-side orchestration
// ---------------------------------------------------------------------------
extern "C" void kernel_launch(void* const* d_in, const int* in_sizes, int n_in,
                              void* d_out, int out_size, void* d_ws, size_t ws_size,
                              hipStream_t stream) {
  (void)in_sizes; (void)n_in; (void)out_size; (void)ws_size;
  const float* hidden = (const float*)d_in[0];
  const float* wq_a   = (const float*)d_in[1];
  const float* qn_w   = (const float*)d_in[2];
  const float* wq_b   = (const float*)d_in[3];
  const float* wkv_a  = (const float*)d_in[4];
  const float* kvn_w  = (const float*)d_in[5];
  const float* wkv_b  = (const float*)d_in[6];
  const float* wo     = (const float*)d_in[7];
  const float* iwqb   = (const float*)d_in[8];
  const float* iwk    = (const float*)d_in[9];
  const float* ikn_w  = (const float*)d_in[10];
  const float* ikn_b  = (const float*)d_in[11];
  const float* iwp    = (const float*)d_in[12];
  const float* fcos   = (const float*)d_in[13];
  const float* fsin   = (const float*)d_in[14];
  float* outp = (float*)d_out;

  // workspace carve-up (256B aligned)
  char* base = (char*)d_ws;
  size_t off = 0;
  auto alloc = [&](size_t bytes) {
    void* p = base + off;
    off += (bytes + 255) & ~(size_t)255;
    return p;
  };
  __bf16* hid_bf  = (__bf16*)alloc((size_t)S * DM * 2);
  __bf16* wqa_bf  = (__bf16*)alloc((size_t)QR * DM * 2);
  __bf16* wqb_bf  = (__bf16*)alloc((size_t)H * DQK * QR * 2);
  __bf16* wkva_bf = (__bf16*)alloc((size_t)(KR + DR) * DM * 2);
  __bf16* wkvb_bf = (__bf16*)alloc((size_t)H * 256 * KR * 2);
  __bf16* wo_bf   = (__bf16*)alloc((size_t)DM * (H * DV) * 2);
  __bf16* iwqb_bf = (__bf16*)alloc((size_t)HI * DI * QR * 2);
  __bf16* iwk_bf  = (__bf16*)alloc((size_t)DI * DM * 2);
  __bf16* iwp_bf  = (__bf16*)alloc((size_t)HI * DM * 2);
  float*  qr32    = (float*)alloc((size_t)S * QR * 4);
  __bf16* qr_bf   = (__bf16*)alloc((size_t)S * QR * 2);
  float*  q32     = (float*)alloc((size_t)S * H * DQK * 4);
  __bf16* q_bf    = (__bf16*)alloc((size_t)S * H * DQK * 2);
  float*  kvall32 = (float*)alloc((size_t)S * (KR + DR) * 4);
  __bf16* kv_bf   = (__bf16*)alloc((size_t)S * KR * 2);
  float*  kpe32   = (float*)alloc((size_t)S * DR * 4);
  float*  kvp32   = (float*)alloc((size_t)S * H * 256 * 4);
  __bf16* k_bf    = (__bf16*)alloc((size_t)S * H * DQK * 2);
  __bf16* vt_bf   = (__bf16*)alloc((size_t)H * DV * S * 2);
  float*  qi32    = (float*)alloc((size_t)S * HI * DI * 4);
  __bf16* qi_bf   = (__bf16*)alloc((size_t)S * HI * DI * 2);
  float*  ki32    = (float*)alloc((size_t)S * DI * 4);
  __bf16* ki_bf   = (__bf16*)alloc((size_t)S * DI * 2);
  float*  wts32   = (float*)alloc((size_t)S * HI * 4);
  float*  isc     = (float*)alloc((size_t)S * S * 4);
  unsigned* maskw = (unsigned*)alloc((size_t)S * 64 * 4);
  __bf16* attn_bf = (__bf16*)alloc((size_t)S * H * DV * 2);

  auto cvt = [&](const float* src, __bf16* dst, size_t n) {
    int blocks = (int)((n + 255) / 256);
    if (blocks > 2048) blocks = 2048;
    cvt_bf16_kernel<<<blocks, 256, 0, stream>>>(src, dst, n);
  };
  cvt(hidden, hid_bf,  (size_t)S * DM);
  cvt(wq_a,   wqa_bf,  (size_t)QR * DM);
  cvt(wq_b,   wqb_bf,  (size_t)H * DQK * QR);
  cvt(wkv_a,  wkva_bf, (size_t)(KR + DR) * DM);
  cvt(wkv_b,  wkvb_bf, (size_t)H * 256 * KR);
  cvt(wo,     wo_bf,   (size_t)DM * H * DV);
  cvt(iwqb,   iwqb_bf, (size_t)HI * DI * QR);
  cvt(iwk,    iwk_bf,  (size_t)DI * DM);
  cvt(iwp,    iwp_bf,  (size_t)HI * DM);

  // TDM/LDS-tiled GEMM for 128-divisible shapes
  auto gemmL = [&](const __bf16* A, const __bf16* B, float* C,
                   int M, int N, int K, float scale) {
    gemm_lds_kernel<<<(M / 128) * (N / 128), 256, 0, stream>>>(A, B, C, M, N, K, scale);
  };
  // direct register-fragment GEMM for the rest
  auto gemm4 = [&](const __bf16* A, const __bf16* B, float* C,
                   int M, int N, int K, float scale) {
    int waves = (M / 16) * (N / 64);
    gemm_bt_kernel<4><<<(waves + 7) / 8, 256, 0, stream>>>(A, B, C, M, N, K, scale);
  };

  // ---- MLA q path
  gemmL(hid_bf, wqa_bf, qr32, S, QR, DM, 1.0f);
  rmsnorm_kernel<<<S, 256, 0, stream>>>(qr32, qn_w, qr_bf, QR, QR, QR);
  gemmL(qr_bf, wqb_bf, q32, S, H * DQK, QR, 1.0f);
  rope_q_kernel<<<2048, 256, 0, stream>>>(q32, fcos, fsin, q_bf);

  // ---- MLA kv path
  gemm4(hid_bf, wkva_bf, kvall32, S, KR + DR, DM, 1.0f);   // N=576
  rmsnorm_kernel<<<S, 256, 0, stream>>>(kvall32, kvn_w, kv_bf, KR, KR + DR, KR);
  kpe_kernel<<<512, 256, 0, stream>>>(kvall32, fcos, fsin, kpe32);
  gemmL(kv_bf, wkvb_bf, kvp32, S, H * 256, KR, 1.0f);
  assemble_kv_kernel<<<2048, 256, 0, stream>>>(kvp32, kpe32, k_bf, vt_bf);

  // ---- indexer path
  gemmL(qr_bf, iwqb_bf, qi32, S, HI * DI, QR, 1.0f);
  rope_qi_kernel<<<2048, 256, 0, stream>>>(qi32, fcos, fsin, qi_bf);
  gemmL(hid_bf, iwk_bf, ki32, S, DI, DM, 1.0f);
  ki_ln_rope_kernel<<<S, 128, 0, stream>>>(ki32, ikn_w, ikn_b, fcos, fsin, ki_bf);
  {
    int waves = (S / 16) * (HI / 32);   // NT=2 -> 32 cols per wave
    gemm_bt_kernel<2><<<(waves + 7) / 8, 256, 0, stream>>>(
        hid_bf, iwp_bf, wts32, S, HI, DM, SC_WTS);
  }
  idx_scores_kernel<<<(S / 16) * (S / 16) / 8, 256, 0, stream>>>(qi_bf, ki_bf, wts32, isc);
  topk_mask_kernel<<<S, 256, 0, stream>>>(isc, maskw);

  // ---- sparse attention + output projection
  attn_kernel<<<S / 16, 512, 0, stream>>>(q_bf, k_bf, vt_bf, maskw, attn_bf);
  gemmL(attn_bf, wo_bf, outp, S, DM, H * DV, 1.0f);
}
